// CausalSelfAttention_70068096466953
// MI455X (gfx1250) — matrix operand order repaired and checked by reading, not compiled, not run on hardware
//
#include <hip/hip_runtime.h>

// ---------------- types ----------------
typedef __attribute__((ext_vector_type(16))) __bf16 v16bf;
typedef __attribute__((ext_vector_type(8)))  __bf16 v8bf;
typedef __attribute__((ext_vector_type(8)))  float  v8f;
typedef __attribute__((ext_vector_type(4)))  unsigned int u32x4;
typedef __attribute__((ext_vector_type(8)))  int i32x8;
typedef __attribute__((ext_vector_type(4)))  int i32x4;

#define T_SEQ 2048
#define NH    16
#define HD    64
#define CDIM  1024
#define BATCH 2
#define MROWS (BATCH * T_SEQ)   // 4096

// ---------------- bf16 helpers ----------------
__device__ __forceinline__ unsigned short f2bf(float f) {
  union { float f; unsigned int u; } v; v.f = f;
  unsigned int r = v.u + 0x7FFFu + ((v.u >> 16) & 1u);   // RNE
  return (unsigned short)(r >> 16);
}
__device__ __forceinline__ float bf2f(unsigned short h) {
  union { unsigned int u; float f; } v; v.u = ((unsigned int)h) << 16;
  return v.f;
}

// ---------------- WMMA fragment loaders (contiguous k = 16*half + s, same order for A and B) ----------------
// A-matrix 16x32 bf16: lane<16 -> row=lane, K = sel*16 + 0..15 contiguous
template <typename PT>
__device__ __forceinline__ v16bf load_a16x32(PT base, int lda, int lane) {
  int row = lane & 15;
  int sel = (lane >> 4) & 1;
  return *(const v16bf*)(base + row * lda + sel * 16);
}
// B-matrix 32x16 bf16 from row-major R[n][k] (B = R^T): lane=n, K = sel*16 + 0..15 contiguous
template <typename PT>
__device__ __forceinline__ v16bf load_b32x16(PT base, int lda, int lane) {
  int col = lane & 15;
  int sel = (lane >> 4) & 1;
  return *(const v16bf*)(base + col * lda + sel * 16);
}

// ---------------- TDM: stage a 64(N-rows) x 32(K) bf16 weight slab into LDS ----------
// 2D tile, data_size=2B: tile_dim0=32 (contiguous k), tile_dim1=64 (weight rows),
// tensor_dim0 = tensor_dim0_stride = CDIM, tensor_dim1 = CDIM. LDS dest is packed
// row-major [64][32] (64B rows, no padding).
__device__ __forceinline__ void tdm_load_b_slab(const unsigned short* W, int n0, int k0,
                                                unsigned lds_off) {
  unsigned long long ga =
      (unsigned long long)W + ((unsigned long long)n0 * CDIM + (unsigned long long)k0) * 2ull;
  u32x4 g0;
  g0[0] = 1u;                                    // count=1, user descriptor
  g0[1] = lds_off;                               // lds_addr (bytes)
  g0[2] = (unsigned)ga;                          // global_addr[31:0]
  g0[3] = (unsigned)(ga >> 32) | (2u << 30);     // global_addr[56:32] | type=2 (image)
  i32x8 g1;
  g1[0] = (int)(1u << 16);                       // workgroup_mask=0, data_size=1 (2 bytes)
  g1[1] = (int)(((unsigned)CDIM & 0xFFFFu) << 16);              // tensor_dim0[15:0]
  g1[2] = (int)(((unsigned)CDIM >> 16) | (((unsigned)CDIM & 0xFFFFu) << 16)); // dim0 hi | dim1 lo
  g1[3] = (int)(((unsigned)CDIM >> 16) | (32u << 16));          // dim1 hi | tile_dim0=32
  g1[4] = 64;                                    // tile_dim1=64, tile_dim2=0
  g1[5] = (int)CDIM;                             // tensor_dim0_stride[31:0]
  g1[6] = 0;                                     // stride0 hi | stride1 lo
  g1[7] = 0;                                     // stride1 hi
  i32x4 z4; z4[0] = 0; z4[1] = 0; z4[2] = 0; z4[3] = 0;
#if __clang_major__ >= 23
  i32x8 z8; z8[0]=0; z8[1]=0; z8[2]=0; z8[3]=0; z8[4]=0; z8[5]=0; z8[6]=0; z8[7]=0;
  __builtin_amdgcn_tensor_load_to_lds(g0, g1, z4, z4, z8, 0);
#else
  __builtin_amdgcn_tensor_load_to_lds(g0, g1, z4, z4, 0);
#endif
}

// ---------------- fp32 -> bf16 pack ----------------
__global__ void cvt_bf16_kernel(const float* __restrict__ in,
                                unsigned short* __restrict__ out, int n) {
  int i = blockIdx.x * blockDim.x + threadIdx.x;
  if (i < n) out[i] = f2bf(in[i]);
}

// ---------------- GEMM: out = A @ W^T + bias ----------------
// Block = 4 waves = 64 M-rows x 64 N-cols. B slab TDM-staged into LDS, double-buffered.
// mode 0: fp32 out[M, CDIM]
// mode 1: bf16 out head-transposed  [B, H, T, HD]
// mode 2: bf16 out head+dim transposed [B, H, HD, T]   (for V)
__global__ __launch_bounds__(128) void gemm_bf16_kernel(
    const unsigned short* __restrict__ A,
    const unsigned short* __restrict__ W,
    const float* __restrict__ bias,
    float* __restrict__ outf,
    unsigned short* __restrict__ outb,
    int M, int mode) {
  __shared__ unsigned short Bslab[2][64 * 32];   // 2 x 4 KB
  int lane = threadIdx.x & 31;
  int wave = threadIdx.x >> 5;
  int ntn  = CDIM / 64;                          // 16 N-blocks
  int bm   = blockIdx.x / ntn;
  int bn   = blockIdx.x % ntn;
  int m0   = bm * 64 + wave * 16;
  int n0   = bn * 64;
  if (m0 >= M) return;                           // never taken (M=4096), uniform anyway

  unsigned lds0 = (unsigned)(unsigned long long)&Bslab[0][0];
  unsigned lds1 = (unsigned)(unsigned long long)&Bslab[1][0];

  v8f acc[4];
#pragma unroll
  for (int t = 0; t < 4; ++t)
#pragma unroll
    for (int j = 0; j < 8; ++j) acc[t][j] = 0.0f;

  // stage first slab
  if (wave == 0) {
    tdm_load_b_slab(W, n0, 0, lds0);
    __builtin_amdgcn_s_wait_tensorcnt(0);
  }
  __syncthreads();

  v16bf a_cur = load_a16x32(A + (size_t)m0 * CDIM, CDIM, lane);

  for (int k0 = 0; k0 < CDIM; k0 += 32) {
    int cur = (k0 >> 5) & 1;
    // kick off next slab DMA while computing on current one
    if (wave == 0 && (k0 + 32) < CDIM)
      tdm_load_b_slab(W, n0, k0 + 32, cur ? lds0 : lds1);
    // register double-buffer of the A fragment
    v16bf a_next = a_cur;
    if ((k0 + 32) < CDIM)
      a_next = load_a16x32(A + (size_t)m0 * CDIM + k0 + 32, CDIM, lane);

    const unsigned short* slab = &Bslab[cur][0];
#pragma unroll
    for (int t = 0; t < 4; ++t) {
      v16bf b = load_b32x16(slab + t * 16 * 32, 32, lane);   // ds_load path
      acc[t] = __builtin_amdgcn_wmma_f32_16x16x32_bf16(
          false, a_cur, false, b, (short)0, acc[t], false, false);
    }
    a_cur = a_next;
    if (wave == 0) __builtin_amdgcn_s_wait_tensorcnt(0);
    __syncthreads();
  }

  int colid = lane & 15;
  int rsel  = (lane & 16) ? 8 : 0;
#pragma unroll
  for (int t = 0; t < 4; ++t) {
    int n = n0 + t * 16 + colid;
    float bv = bias[n];
#pragma unroll
    for (int j = 0; j < 8; ++j) {
      int row = m0 + j + rsel;
      float val = acc[t][j] + bv;
      if (mode == 0) {
        outf[(size_t)row * CDIM + n] = val;
      } else {
        int b = row / T_SEQ, tt = row % T_SEQ;
        int h = n >> 6, d = n & 63;
        if (mode == 1)
          outb[(((size_t)(b * NH + h) * T_SEQ) + tt) * HD + d] = f2bf(val);
        else
          outb[(((size_t)(b * NH + h) * HD) + d) * T_SEQ + tt] = f2bf(val);
      }
    }
  }
}

// ---------------- in-place RoPE on bf16 [B,H,T,HD] ----------------
__global__ void rope_kernel(unsigned short* __restrict__ Qb, int total) {
  int idx = blockIdx.x * blockDim.x + threadIdx.x;
  if (idx >= total) return;                  // total = B*H*T*(HD/2)
  int j  = idx & 31;                         // rotary pair index
  int t  = (idx >> 5) & (T_SEQ - 1);
  int bh = idx >> 16;                        // 2048*32 = 65536 per (b,h)
  unsigned short* p = Qb + ((size_t)(bh * T_SEQ) + t) * HD + 2 * j;
  float x0 = bf2f(p[0]), x1 = bf2f(p[1]);
  float inv = __expf(-((float)j / 32.0f) * 9.210340371976184f);  // 10000^(-j/32)
  float ang = (float)t * inv;
  float s, c;
  __sincosf(ang, &s, &c);
  p[0] = f2bf(c * x0 - s * x1);
  p[1] = f2bf(s * x0 + c * x1);
}

// ---------------- flash attention: one wave per 16-row Q tile ----------------
__global__ __launch_bounds__(128) void attn_kernel(
    const unsigned short* __restrict__ Qb,   // [B,H,T,HD] bf16 (rope'd)
    const unsigned short* __restrict__ Kb,   // [B,H,T,HD] bf16 (rope'd)
    const unsigned short* __restrict__ Vt,   // [B,H,HD,T] bf16
    unsigned short* __restrict__ Yb) {       // [B*T, CDIM] bf16
  __shared__ unsigned short Pbuf[4][16 * 32];
  int lane = threadIdx.x & 31;
  int wave = threadIdx.x >> 5;
  int tile = blockIdx.x * 4 + wave;          // B*H*(T/16) = 4096 tiles
  int qt = tile & (T_SEQ / 16 - 1);
  int bh = tile >> 7;
  int qbase = qt * 16;
  const unsigned short* Qh = Qb + (size_t)bh * T_SEQ * HD;
  const unsigned short* Kh = Kb + (size_t)bh * T_SEQ * HD;
  const unsigned short* Vh = Vt + (size_t)bh * HD * T_SEQ;
  unsigned short* P = Pbuf[wave];

  // Q fragments over the two hd K-slices
  v16bf qa0 = load_a16x32(Qh + qbase * HD + 0, HD, lane);
  v16bf qa1 = load_a16x32(Qh + qbase * HD + 32, HD, lane);

  v8f o[4];
  float mrow[8], lrow[8];
#pragma unroll
  for (int t = 0; t < 4; ++t)
#pragma unroll
    for (int j = 0; j < 8; ++j) o[t][j] = 0.0f;
#pragma unroll
  for (int j = 0; j < 8; ++j) { mrow[j] = -3.0e38f; lrow[j] = 0.0f; }

  int colid = lane & 15;
  int rsel  = (lane & 16) ? 8 : 0;
  const float scale = 0.125f;                // 1/sqrt(64)
  int jend = qbase + 16;                     // causal bound (exclusive)

  for (int j0 = 0; j0 < jend; j0 += 32) {
    // S = Q K^T for a 16x32 KV slab (two 16x16 N-tiles)
    v8f s0, s1;
#pragma unroll
    for (int j = 0; j < 8; ++j) { s0[j] = 0.0f; s1[j] = 0.0f; }
    {
      v16bf b = load_b32x16(Kh + (size_t)j0 * HD + 0, HD, lane);
      s0 = __builtin_amdgcn_wmma_f32_16x16x32_bf16(false, qa0, false, b, (short)0, s0, false, false);
      b = load_b32x16(Kh + (size_t)j0 * HD + 32, HD, lane);
      s0 = __builtin_amdgcn_wmma_f32_16x16x32_bf16(false, qa1, false, b, (short)0, s0, false, false);
      b = load_b32x16(Kh + (size_t)(j0 + 16) * HD + 0, HD, lane);
      s1 = __builtin_amdgcn_wmma_f32_16x16x32_bf16(false, qa0, false, b, (short)0, s1, false, false);
      b = load_b32x16(Kh + (size_t)(j0 + 16) * HD + 32, HD, lane);
      s1 = __builtin_amdgcn_wmma_f32_16x16x32_bf16(false, qa1, false, b, (short)0, s1, false, false);
    }
    // scale + causal mask + online softmax (row = per-VGPR, 16-lane half-wave reduce)
#pragma unroll
    for (int j = 0; j < 8; ++j) {
      int rg = qbase + j + rsel;
      float v0 = s0[j] * scale;
      float v1 = s1[j] * scale;
      if (j0 + colid > rg)      v0 = -1.0e30f;
      if (j0 + 16 + colid > rg) v1 = -1.0e30f;
      float mx = fmaxf(v0, v1);
#pragma unroll
      for (int off = 1; off < 16; off <<= 1)
        mx = fmaxf(mx, __shfl_xor(mx, off, 32));
      float mold = mrow[j];
      float mnew = fmaxf(mold, mx);
      float alpha = __expf(mold - mnew);
      float p0 = __expf(v0 - mnew);
      float p1 = __expf(v1 - mnew);
      float rs = p0 + p1;
#pragma unroll
      for (int off = 1; off < 16; off <<= 1)
        rs += __shfl_xor(rs, off, 32);
      mrow[j] = mnew;
      lrow[j] = lrow[j] * alpha + rs;
      o[0][j] *= alpha; o[1][j] *= alpha; o[2][j] *= alpha; o[3][j] *= alpha;
      int r = j + rsel;
      P[r * 32 + colid]      = f2bf(p0);
      P[r * 32 + 16 + colid] = f2bf(p1);
    }
    // P (C/D-layout) was staged in LDS; re-read in A-fragment layout
    asm volatile("s_wait_dscnt 0" ::: "memory");
    v16bf pa = load_a16x32((const unsigned short*)P, 32, lane);
#pragma unroll
    for (int t = 0; t < 4; ++t) {
      v16bf vb = load_b32x16(Vh + (size_t)(t * 16) * T_SEQ + j0, T_SEQ, lane);
      o[t] = __builtin_amdgcn_wmma_f32_16x16x32_bf16(
          false, pa, false, vb, (short)0, o[t], false, false);
    }
  }

  int b = bh >> 4, h = bh & 15;
#pragma unroll
  for (int t = 0; t < 4; ++t) {
#pragma unroll
    for (int j = 0; j < 8; ++j) {
      int rg = qbase + j + rsel;
      float val = o[t][j] / lrow[j];
      size_t row = (size_t)b * T_SEQ + rg;
      int n = h * HD + t * 16 + colid;
      Yb[row * CDIM + n] = f2bf(val);
    }
  }
}

// ---------------- host side ----------------
extern "C" void kernel_launch(void* const* d_in, const int* in_sizes, int n_in,
                              void* d_out, int out_size, void* d_ws, size_t ws_size,
                              hipStream_t stream) {
  (void)in_sizes; (void)n_in; (void)out_size; (void)ws_size;
  const float* x  = (const float*)d_in[0];
  const float* Wq = (const float*)d_in[1];
  const float* bq = (const float*)d_in[2];
  const float* Wk = (const float*)d_in[3];
  const float* bk = (const float*)d_in[4];
  const float* Wv = (const float*)d_in[5];
  const float* bv = (const float*)d_in[6];
  const float* Wp = (const float*)d_in[7];
  const float* bp = (const float*)d_in[8];
  float* out = (float*)d_out;

  // workspace carve-up (48 MB total, all 2MB-aligned offsets)
  unsigned short* xb  = (unsigned short*)d_ws;            // 4096*1024
  unsigned short* wqb = xb  + (size_t)MROWS * CDIM;       // 1024*1024 each
  unsigned short* wkb = wqb + (size_t)CDIM * CDIM;
  unsigned short* wvb = wkb + (size_t)CDIM * CDIM;
  unsigned short* wpb = wvb + (size_t)CDIM * CDIM;
  unsigned short* Qb  = wpb + (size_t)CDIM * CDIM;        // [B,H,T,HD]
  unsigned short* Kb  = Qb  + (size_t)MROWS * CDIM;
  unsigned short* Vtb = Kb  + (size_t)MROWS * CDIM;       // [B,H,HD,T]
  unsigned short* yb  = Vtb + (size_t)MROWS * CDIM;       // [B*T, C]

  const int nX = MROWS * CDIM;    // 4194304
  const int nW = CDIM * CDIM;     // 1048576

  cvt_bf16_kernel<<<nX / 256, 256, 0, stream>>>(x,  xb,  nX);
  cvt_bf16_kernel<<<nW / 256, 256, 0, stream>>>(Wq, wqb, nW);
  cvt_bf16_kernel<<<nW / 256, 256, 0, stream>>>(Wk, wkb, nW);
  cvt_bf16_kernel<<<nW / 256, 256, 0, stream>>>(Wv, wvb, nW);
  cvt_bf16_kernel<<<nW / 256, 256, 0, stream>>>(Wp, wpb, nW);

  dim3 ggrid((MROWS / 64) * (CDIM / 64));       // 1024 blocks, 4 waves each
  dim3 gblk(128);
  gemm_bf16_kernel<<<ggrid, gblk, 0, stream>>>(xb, wqb, bq, nullptr, Qb,  MROWS, 1);
  gemm_bf16_kernel<<<ggrid, gblk, 0, stream>>>(xb, wkb, bk, nullptr, Kb,  MROWS, 1);
  gemm_bf16_kernel<<<ggrid, gblk, 0, stream>>>(xb, wvb, bv, nullptr, Vtb, MROWS, 2);

  const int nRope = BATCH * NH * T_SEQ * (HD / 2);  // 2097152
  rope_kernel<<<nRope / 256, 256, 0, stream>>>(Qb, nRope);
  rope_kernel<<<nRope / 256, 256, 0, stream>>>(Kb, nRope);

  dim3 agrid(BATCH * NH * (T_SEQ / 16) / 4);    // 1024 blocks, 4 waves each
  attn_kernel<<<agrid, dim3(128), 0, stream>>>(Qb, Kb, Vtb, yb);

  gemm_bf16_kernel<<<ggrid, gblk, 0, stream>>>(yb, wpb, bp, out, nullptr, MROWS, 0);
}
